// StridedPattern_55490977465136
// MI455X (gfx1250) — matrix operand order repaired and checked by reading, not compiled, not run on hardware
//
#include <hip/hip_runtime.h>

typedef __attribute__((ext_vector_type(16))) __bf16 v16bf;
typedef __attribute__((ext_vector_type(8)))  float  v8f;
typedef __attribute__((ext_vector_type(4)))  float  f4;
typedef __attribute__((ext_vector_type(4)))  int    v4i;

#define B_ 4
#define S_ 4096
#define D_ 1024
#define E_ 64
#define M_ 1024
#define STRIDE_ 4

#if __has_builtin(__builtin_amdgcn_global_load_async_to_lds_b128) && \
    __has_builtin(__builtin_amdgcn_s_wait_asynccnt)
#define HAVE_ASYNC_LDS 1
#else
#define HAVE_ASYNC_LDS 0
#endif

__device__ __forceinline__ int wave_sum_i32(int x) {
#pragma unroll
  for (int o = 16; o > 0; o >>= 1) x += __shfl_xor(x, o, 32);
  return x;
}

__device__ __forceinline__ void pack4(v16bf& dst, int at, float4 f) {
  dst[at + 0] = (__bf16)f.x; dst[at + 1] = (__bf16)f.y;
  dst[at + 2] = (__bf16)f.z; dst[at + 3] = (__bf16)f.w;
}

// ---------------- Projection GEMM (f32 in, bf16 out) --------------------------------
// Y[b,r,e] = sum_d X[b, r*rowStride, d] * W[e,d].  One wave per 16x16 output tile,
// K=1024 in 32-wide steps via v_wmma_f32_16x16x32_bf16; f32->bf16 conversion done
// in-register so x is read once in f32 (repeat column-tile reads are served by L2).
__global__ void proj_gemm(const float* __restrict__ X, const float* __restrict__ W,
                          __bf16* __restrict__ Y, int rows, int rowStride) {
  int wave = (blockIdx.x * blockDim.x + threadIdx.x) >> 5;
  int lane = threadIdx.x & 31;
  int tilesPerBatch = (rows >> 4) * (E_ >> 4);
  if (wave >= B_ * tilesPerBatch) return;
  int b  = wave / tilesPerBatch;
  int t  = wave - b * tilesPerBatch;
  int rt = t >> 2;   // 16-row tile
  int ct = t & 3;    // 16-col tile (of E=64)
  int half = lane >> 4;
  int lr   = lane & 15;

  const float* arow = X + (size_t)b * S_ * D_ + (size_t)((rt * 16 + lr) * rowStride) * D_;
  const float* brow = W + (size_t)(ct * 16 + lr) * D_;

  v8f c = {};
#pragma unroll 4
  for (int k = 0; k < D_; k += 32) {
    // A 16x32 lane layout: lane<16 holds K {k..k+7, k+16..k+23}; lanes>=16 shifted by 8
    float4 a0 = *(const float4*)(arow + k + half * 8);
    float4 a1 = *(const float4*)(arow + k + half * 8 + 4);
    float4 a2 = *(const float4*)(arow + k + 16 + half * 8);
    float4 a3 = *(const float4*)(arow + k + 16 + half * 8 + 4);
    // B 32x16 lane layout: lane<16 holds K {k..k+7, k+16..k+23} of col lr; lanes>=16 shifted by 8
    float4 b0 = *(const float4*)(brow + k + half * 8);
    float4 b1 = *(const float4*)(brow + k + half * 8 + 4);
    float4 b2 = *(const float4*)(brow + k + 16 + half * 8);
    float4 b3 = *(const float4*)(brow + k + 16 + half * 8 + 4);
    if (k + 32 < D_) __builtin_prefetch(arow + k + 128, 0, 3);
    v16bf a, bm;
    pack4(a, 0, a0);  pack4(a, 4, a1);  pack4(a, 8, a2);  pack4(a, 12, a3);
    pack4(bm, 0, b0); pack4(bm, 4, b1); pack4(bm, 8, b2); pack4(bm, 12, b3);
    c = __builtin_amdgcn_wmma_f32_16x16x32_bf16(false, a, false, bm,
                                                (short)0, c, false, false);
  }
  __bf16* yb = Y + (size_t)b * rows * E_;
#pragma unroll
  for (int r = 0; r < 8; ++r) {
    int row = rt * 16 + r + 8 * half;     // C layout: VGPR r holds rows r / r+8
    yb[(size_t)row * E_ + ct * 16 + lr] = (__bf16)c[r];
  }
}

// ---------------- Fused score GEMM + top-k selection + mask write -------------------
// One workgroup (16 waves / 512 threads) per (batch, 16-query-row stripe).
// Stage 0: async-copy the 2 KB Q stripe to LDS once (ASYNCcnt path); all waves then
//          read A-fragments from LDS instead of 16x-redundant global fetches.
// Stage 1: wave w computes score cols [64w, 64w+64) via 4 WMMA tiles (K=64),
//          relu -> LDS scores (16x1024 f32 = 64 KB of the WGP's 320 KB).
// Stage 2: barrier; wave w selects top-kk for row w from LDS (31-step binary search
//          on IEEE bits, stable tie-break) and streams the 16 KB output row with
//          coalesced NON-TEMPORAL float4 stores (write-once stream must not rinse L2).
__global__ __launch_bounds__(512) void score_select(const __bf16* __restrict__ Q,
                                                    const __bf16* __restrict__ K,
                                                    float* __restrict__ out) {
  __shared__ float  sc[16][M_];           // 64 KB score stripe
  __shared__ __bf16 qtile[16 * E_];       // 2 KB Q stripe

  int wg = blockIdx.x;
  if (wg >= B_ * (S_ >> 4)) return;       // uniform per block
  int b  = wg >> 8;                       // 256 stripes per batch
  int rt = wg & 255;
  int s0 = rt * 16;

  int tid  = threadIdx.x;
  int w    = tid >> 5;                    // wave 0..15
  int lane = tid & 31;
  int half = lane >> 4;
  int lr   = lane & 15;

  // ---- Stage 0: Q stripe (1024 bf16 = 2 KB, contiguous) -> LDS ----
  const __bf16* qsrc = Q + ((size_t)b * S_ + s0) * E_;
#if HAVE_ASYNC_LDS
  if (tid < 128) {
    __builtin_amdgcn_global_load_async_to_lds_b128(
        (__attribute__((address_space(1))) v4i*)(qsrc + tid * 8),
        (__attribute__((address_space(3))) v4i*)(qtile + tid * 8),
        0, 0);
  }
  __builtin_amdgcn_s_wait_asynccnt(0);
#else
  if (tid < 128)
    *(float4*)&qtile[tid * 8] = *(const float4*)(qsrc + tid * 8);
#endif
  __syncthreads();

  // ---- Stage 1: 4 column tiles per wave, A-fragments from LDS ----
  const __bf16* qrow = qtile + lr * E_;
#pragma unroll
  for (int ct4 = 0; ct4 < 4; ++ct4) {
    int ct = w * 4 + ct4;                 // 16-col tile index, 0..63
    const __bf16* krow = K + ((size_t)b * M_ + ct * 16 + lr) * E_;
    v8f c = {};
#pragma unroll
    for (int k = 0; k < E_; k += 32) {
      v16bf a, bm;
      *(float4*)&a         = *(const float4*)(qrow + k + half * 16);      // ds_load
      *(((float4*)&a) + 1) = *(const float4*)(qrow + k + half * 16 + 8);  // ds_load
      bm = *(const v16bf*)(krow + k + half * 16);                          // global
      c = __builtin_amdgcn_wmma_f32_16x16x32_bf16(false, a, false, bm,
                                                  (short)0, c, false, false);
    }
#pragma unroll
    for (int r = 0; r < 8; ++r) {
      int row = r + 8 * half;
      float v = c[r];
      sc[row][ct * 16 + lr] = (v > 0.0f) ? v : 0.0f;   // relu, force +0
    }
  }
  __syncthreads();

  // ---- Stage 2: wave w owns query row s0 + w ----
  int s  = s0 + w;
  int n  = (s >> 2) + 1;                  // valid strided keys
  int kk = n >> 1; if (kk < 1) kk = 1;

  float v[32];
  const float4* sp = (const float4*)&sc[w][lane * 32];
#pragma unroll
  for (int j = 0; j < 8; ++j) {
    float4 t = sp[j];
    v[4 * j] = t.x; v[4 * j + 1] = t.y; v[4 * j + 2] = t.z; v[4 * j + 3] = t.w;
  }
  int base = lane * 32;
  int nv = n - base; nv = nv < 0 ? 0 : (nv > 32 ? 32 : nv);

  // kk-th largest = max t with count(key >= t) >= kk (bits monotone: all vals >= +0)
  unsigned prefix = 0u;
  for (int bit = 30; bit >= 0; --bit) {
    unsigned cand = prefix | (1u << bit);
    int c = 0;
#pragma unroll
    for (int j = 0; j < 32; ++j)
      c += (j < nv && __float_as_uint(v[j]) >= cand) ? 1 : 0;
    if (wave_sum_i32(c) >= kk) prefix = cand;
  }
  const unsigned T = prefix;

  int cgt = 0, ceq = 0;
#pragma unroll
  for (int j = 0; j < 32; ++j) {
    unsigned kj = __float_as_uint(v[j]);
    cgt += (j < nv && kj > T) ? 1 : 0;
    ceq += (j < nv && kj == T) ? 1 : 0;
  }
  int totalGt = wave_sum_i32(cgt);
  int inc = ceq;                          // exclusive prefix of tie counts by index
#pragma unroll
  for (int o = 1; o < 32; o <<= 1) {
    int y = __shfl_up(inc, o, 32);
    if (lane >= o) inc += y;
  }
  int run   = inc - ceq;
  int extra = kk - totalGt;

  unsigned mask = 0u;
#pragma unroll
  for (int j = 0; j < 32; ++j) {
    bool sel = false;
    if (j < nv) {
      unsigned kj = __float_as_uint(v[j]);
      if (kj > T) sel = true;
      else if (kj == T) { if (run < extra) sel = true; run++; }
    }
    mask |= sel ? (1u << j) : 0u;
  }

  const float NEG = -__builtin_inff();
  f4* orow = (f4*)(out + ((size_t)b * S_ + s) * S_);
#pragma unroll 4
  for (int i = 0; i < 32; ++i) {
    unsigned mi = __shfl((int)mask, i, 32);  // lane i owns strided cols [128i, 128i+128)
    f4 val;
    val.x = ((mi >> lane) & 1u) ? 0.0f : NEG;
    val.y = NEG; val.z = NEG; val.w = NEG;
    __builtin_nontemporal_store(val, orow + i * 32 + lane);  // TH=NT: don't rinse L2
  }
}

extern "C" void kernel_launch(void* const* d_in, const int* in_sizes, int n_in,
                              void* d_out, int out_size, void* d_ws, size_t ws_size,
                              hipStream_t stream) {
  const float* x  = (const float*)d_in[0];
  const float* wq = (const float*)d_in[1];
  const float* wk = (const float*)d_in[2];
  // d_in[3] = stride scalar (fixed at 4; shapes compiled in)
  float* out = (float*)d_out;

  char* ws = (char*)d_ws;
  size_t off = 0;
  __bf16* Qb = (__bf16*)(ws + off); off += (size_t)B_ * S_ * E_ * 2;   // 2 MiB
  __bf16* Kb = (__bf16*)(ws + off); off += (size_t)B_ * M_ * E_ * 2;   // 0.5 MiB
  (void)ws_size; (void)in_sizes; (void)n_in; (void)out_size;

  // Q projection: 4 * 256 row-tiles * 4 col-tiles = 4096 waves -> 512 blocks of 8 waves
  proj_gemm<<<512, 256, 0, stream>>>(x, wq, Qb, S_, 1);
  // Strided K projection: 4 * 64 * 4 = 1024 waves -> 128 blocks
  proj_gemm<<<128, 256, 0, stream>>>(x, wk, Kb, M_, STRIDE_);

  // Fused scores + selection + 256 MB NT mask write:
  // B * S/16 = 1024 workgroups of 512 threads (16 waves)
  score_select<<<1024, 512, 0, stream>>>(Qb, Kb, out);
}
